// SimCLR_32461362823220
// MI455X (gfx1250) — matrix operand order repaired and checked by reading, not compiled, not run on hardware
//
#include <hip/hip_runtime.h>

// ---------------------------------------------------------------------------
// SimCLR loss on MI455X (gfx1250, wave32, WMMA bf16 16x16x32)
//   B=2048, D_IN=2048, D_OUT=128, TEMP=0.5
//
// All LDS tiles are bf16 in *fragment-native* layout so every WMMA fragment
// is assembled with b128 loads:
//   A-frag lane(L): row m=L%16, K = grp*16..+15                    (2 x b128)
//   B-frag lane(L): col n=L%16, K = grp*16..+15                    (2 x b128)
// Both GEMM kernels double-buffer their LDS tiles (global->reg->LDS) to
// overlap HBM/L2 latency with the WMMA pipe.
// The fused log-sum-exp uses a FIXED max of 2.0: rows are L2-normalized so
// logit = sim/T is bounded by 2, making online max-rescaling unnecessary
// (halves exp count, removes serial dependency, branchless masking).
// ---------------------------------------------------------------------------

typedef __attribute__((ext_vector_type(16))) __bf16 v16bf;
typedef __attribute__((ext_vector_type(8)))  float  v8f;

namespace {
constexpr int kB    = 2048;   // batch
constexpr int kDin  = 2048;   // input dim
constexpr int kDout = 128;    // projection dim
constexpr int kN2   = 4096;   // 2B rows of reps
constexpr float kInvT = 2.0f; // 1/TEMP
}

union FragBF {
    v16bf v;
    uint4 q[2];
    unsigned short u[16];
};

__device__ __forceinline__ unsigned short bf16_bits(float f) {
    union { __bf16 h; unsigned short u; } t;
    t.h = (__bf16)f;
    return t.u;
}

__device__ __forceinline__ unsigned int pack_bf16x2(float lo, float hi) {
    union { __bf16 h[2]; unsigned int u; } t;
    t.h[0] = (__bf16)lo;
    t.h[1] = (__bf16)hi;
    return t.u;
}

// ---------------------------------------------------------------------------
// Kernel 0: Wt[n][k] = bf16(W[k][n])  (one-time transpose+convert, 512 KB)
// ---------------------------------------------------------------------------
__global__ __launch_bounds__(256) void w_transpose_bf16(
    const float* __restrict__ W, unsigned short* __restrict__ Wt)
{
    int tid = blockIdx.x * 256 + threadIdx.x;   // 0..32767
    int k   = tid >> 4;                         // 0..2047
    int n0  = (tid & 15) << 3;                  // 0,8,...,120
    float4 a = *(const float4*)(W + (size_t)k * kDout + n0);
    float4 b = *(const float4*)(W + (size_t)k * kDout + n0 + 4);
#pragma unroll
    for (int j = 0; j < 4; ++j)
        Wt[(size_t)(n0 + j) * kDin + k] = bf16_bits(((const float*)&a)[j]);
#pragma unroll
    for (int j = 0; j < 4; ++j)
        Wt[(size_t)(n0 + 4 + j) * kDin + k] = bf16_bits(((const float*)&b)[j]);
}

// ---------------------------------------------------------------------------
// Kernel 1: z = [x_i; x_j] @ W + b, L2-normalize rows, emit rn (bf16).
// 32 blocks x 256 threads (8 waves); wave -> 16 rows; block -> 128 rows.
// Double-buffered LDS tiles over the K loop (64 steps of 32).
// ---------------------------------------------------------------------------
__global__ __launch_bounds__(256) void simclr_encode(
    const float* __restrict__ xi, const float* __restrict__ xj,
    const unsigned short* __restrict__ Wt, const float* __restrict__ bias,
    unsigned short* __restrict__ rn)
{
    // stride 40 u16 = 80 B: 16B-aligned b128 slots, 20-bank row stride
    __shared__ __align__(16) unsigned short sWt[2][kDout][40];  // [n][k] tile
    __shared__ __align__(16) unsigned short sX[2][8][16][40];   // [m][k] tile

    const int t    = threadIdx.x;
    const int lane = t & 31;
    const int wave = t >> 5;
    const int rowbase = blockIdx.x * 128 + wave * 16;
    const int grp  = lane >> 4;
    const int ncol = lane & 15;

    v8f acc[8];
    const v8f vzero = {0.f,0.f,0.f,0.f,0.f,0.f,0.f,0.f};
#pragma unroll
    for (int nt = 0; nt < 8; ++nt) acc[nt] = vzero;

    // staging indices (fixed per thread)
    const int wn0 = t >> 2, wseg = (t & 3) << 3;        // W tile: 2 slots
    const int wn1 = wn0 + 64;
    const int xr   = lane >> 1;
    const int half = (lane & 1) << 4;
    const int R    = rowbase + xr;
    const float* xsrc = (R < kB) ? (xi + (size_t)R * kDin)
                                 : (xj + (size_t)(R - kB) * kDin);

    uint4  wr0, wr1;
    float4 xv0, xv1, xv2, xv3;

    auto load_tile = [&](int k0) {
        wr0 = *(const uint4*)(Wt + (size_t)wn0 * kDin + k0 + wseg);
        wr1 = *(const uint4*)(Wt + (size_t)wn1 * kDin + k0 + wseg);
        xv0 = *(const float4*)(xsrc + k0 + half);
        xv1 = *(const float4*)(xsrc + k0 + half + 4);
        xv2 = *(const float4*)(xsrc + k0 + half + 8);
        xv3 = *(const float4*)(xsrc + k0 + half + 12);
    };
    auto store_tile = [&](int buf) {
        *(uint4*)(&sWt[buf][wn0][wseg]) = wr0;
        *(uint4*)(&sWt[buf][wn1][wseg]) = wr1;
        uint4 p0, p1;
        p0.x = pack_bf16x2(xv0.x, xv0.y); p0.y = pack_bf16x2(xv0.z, xv0.w);
        p0.z = pack_bf16x2(xv1.x, xv1.y); p0.w = pack_bf16x2(xv1.z, xv1.w);
        p1.x = pack_bf16x2(xv2.x, xv2.y); p1.y = pack_bf16x2(xv2.z, xv2.w);
        p1.z = pack_bf16x2(xv3.x, xv3.y); p1.w = pack_bf16x2(xv3.z, xv3.w);
        *(uint4*)(&sX[buf][wave][xr][half])     = p0;
        *(uint4*)(&sX[buf][wave][xr][half + 8]) = p1;
    };

    // prologue: fill buffer 0
    load_tile(0);
    store_tile(0);
    __syncthreads();

    constexpr int kSteps = kDin / 32;   // 64
    for (int s = 0; s < kSteps; ++s) {
        const int cur = s & 1;
        if (s + 1 < kSteps) load_tile((s + 1) * 32);

        // ---- compute on current buffer
        FragBF afr;
        afr.q[0] = *(const uint4*)(&sX[cur][wave][ncol][grp * 16]);
        afr.q[1] = *(const uint4*)(&sX[cur][wave][ncol][grp * 16 + 8]);
#pragma unroll
        for (int nt = 0; nt < 8; ++nt) {
            FragBF bfr;
            bfr.q[0] = *(const uint4*)(&sWt[cur][nt * 16 + ncol][grp * 16]);
            bfr.q[1] = *(const uint4*)(&sWt[cur][nt * 16 + ncol][grp * 16 + 8]);
            acc[nt] = __builtin_amdgcn_wmma_f32_16x16x32_bf16(
                false, afr.v, false, bfr.v, (short)0, acc[nt], false, false);
        }

        if (s + 1 < kSteps) store_tile(cur ^ 1);
        __syncthreads();
    }

    // ---- epilogue: bias, row norms (cross-lane within 16-lane group), store
    float ss[8];
#pragma unroll
    for (int r = 0; r < 8; ++r) ss[r] = 0.f;
#pragma unroll
    for (int nt = 0; nt < 8; ++nt) {
        float bb = bias[nt * 16 + ncol];
#pragma unroll
        for (int r = 0; r < 8; ++r) {
            float z = acc[nt][r] + bb;
            acc[nt][r] = z;
            ss[r] += z * z;
        }
    }
#pragma unroll
    for (int mask = 1; mask <= 8; mask <<= 1) {
#pragma unroll
        for (int r = 0; r < 8; ++r)
            ss[r] += __shfl_xor(ss[r], mask, 32);
    }
    float inv[8];
#pragma unroll
    for (int r = 0; r < 8; ++r) {
        float nrm = fmaxf(sqrtf(ss[r]), 1e-8f);   // torch-style norm clamp
        inv[r] = 1.0f / nrm;
    }
#pragma unroll
    for (int nt = 0; nt < 8; ++nt) {
#pragma unroll
        for (int r = 0; r < 8; ++r) {
            int Rg = rowbase + (grp << 3) + r;    // C-layout: M = r + 8*grp
            rn[(size_t)Rg * kDout + nt * 16 + ncol] =
                bf16_bits(acc[nt][r] * inv[r]);
        }
    }
}

// ---------------------------------------------------------------------------
// Kernel 2: sim = rn @ rn^T (tiled, never materialized) fused with masked
// fixed-max log-sum-exp (|logit| <= 2 since rows are unit vectors):
//   lse_i = 2 + log( sum_{j != i} exp(logit_ij - 2) )
// Accumulates mean(lse - pos) into *out.
// 64-column panels, double-buffered in LDS.
// ---------------------------------------------------------------------------
__global__ __launch_bounds__(256) void simclr_loss_k(
    const unsigned short* __restrict__ rn, float* __restrict__ out)
{
    __shared__ __align__(16) unsigned short sB[2][64][kDout + 8]; // 2x64x136

    const int t    = threadIdx.x;
    const int lane = t & 31;
    const int wave = t >> 5;
    const int rowbase = blockIdx.x * 128 + wave * 16;
    const int grp  = lane >> 4;
    const int ncol = lane & 15;

    // ---- resident A panel: 16 rows x 128 K, two global b128 loads per frag
    FragBF a[4];
    const unsigned short* arow = rn + (size_t)(rowbase + ncol) * kDout;
#pragma unroll
    for (int c = 0; c < 4; ++c) {
        a[c].q[0] = *(const uint4*)(arow + c * 32 + grp * 16);
        a[c].q[1] = *(const uint4*)(arow + c * 32 + grp * 16 + 8);
    }

    // fixed row / partner-column ids for this lane's 8 rows
    int Mg[8], Pc[8];
#pragma unroll
    for (int r = 0; r < 8; ++r) {
        Mg[r] = rowbase + (grp << 3) + r;
        Pc[r] = (Mg[r] + kB) & (kN2 - 1);
    }
    float s_[8], p_[8];
#pragma unroll
    for (int r = 0; r < 8; ++r) { s_[r] = 0.f; p_[r] = 0.f; }

    // panel staging indices (fixed per thread): 64 rows x 128 cols bf16
    const int pr  = t >> 2;              // 0..63
    const int pseg = (t & 3) << 5;       // 0,32,64,96 (8-u16 granules x4)
    uint4 rb0, rb1, rb2, rb3;

    auto load_panel = [&](int j0) {
        const unsigned short* src = rn + (size_t)(j0 + pr) * kDout + pseg;
        rb0 = *(const uint4*)(src);
        rb1 = *(const uint4*)(src + 8);
        rb2 = *(const uint4*)(src + 16);
        rb3 = *(const uint4*)(src + 24);
    };
    auto store_panel = [&](int buf) {
        *(uint4*)(&sB[buf][pr][pseg])      = rb0;
        *(uint4*)(&sB[buf][pr][pseg + 8])  = rb1;
        *(uint4*)(&sB[buf][pr][pseg + 16]) = rb2;
        *(uint4*)(&sB[buf][pr][pseg + 24]) = rb3;
    };

    const v8f vzero = {0.f,0.f,0.f,0.f,0.f,0.f,0.f,0.f};

    load_panel(0);
    store_panel(0);
    __syncthreads();

    constexpr int kPanels = kN2 / 64;   // 64
    for (int p = 0; p < kPanels; ++p) {
        const int cur = p & 1;
        const int j0  = p * 64;
        if (p + 1 < kPanels) load_panel((p + 1) * 64);

#pragma unroll
        for (int jt = 0; jt < 4; ++jt) {
            v8f acc = vzero;
#pragma unroll
            for (int c = 0; c < 4; ++c) {
                FragBF bfr;   // B[k][n] = rn[j0+jt*16+n][k]
                bfr.q[0] = *(const uint4*)(&sB[cur][jt * 16 + ncol][c * 32 + grp * 16]);
                bfr.q[1] = *(const uint4*)(&sB[cur][jt * 16 + ncol][c * 32 + grp * 16 + 8]);
                acc = __builtin_amdgcn_wmma_f32_16x16x32_bf16(
                    false, a[c].v, false, bfr.v, (short)0, acc, false, false);
            }
            // ---- fused fixed-max LSE update (branchless)
            const int Nc = j0 + jt * 16 + ncol;
#pragma unroll
            for (int r = 0; r < 8; ++r) {
                float logit = acc[r] * kInvT;
                float e = __expf(logit - 2.0f);       // logit <= 2 always
                s_[r] += (Nc == Mg[r]) ? 0.f : e;     // mask diagonal
                p_[r] += (Nc == Pc[r]) ? logit : 0.f; // positive pair
            }
        }

        if (p + 1 < kPanels) store_panel(cur ^ 1);
        __syncthreads();
    }

    // ---- plain sums across the 16-lane group (no max merge needed)
#pragma unroll
    for (int mask = 1; mask <= 8; mask <<= 1) {
#pragma unroll
        for (int r = 0; r < 8; ++r) {
            s_[r] += __shfl_xor(s_[r], mask, 32);
            p_[r] += __shfl_xor(p_[r], mask, 32);
        }
    }

    if (ncol == 0) {   // one lane per 8-row group contributes
        float local = 0.f;
#pragma unroll
        for (int r = 0; r < 8; ++r)
            local += 2.0f + __logf(s_[r]) - p_[r];
        atomicAdd(out, local * (1.0f / (float)kN2));
    }
}

// ---------------------------------------------------------------------------
extern "C" void kernel_launch(void* const* d_in, const int* in_sizes, int n_in,
                              void* d_out, int out_size, void* d_ws, size_t ws_size,
                              hipStream_t stream) {
    const float* x_i  = (const float*)d_in[0];
    const float* x_j  = (const float*)d_in[1];
    const float* W    = (const float*)d_in[2];
    const float* bias = (const float*)d_in[3];
    float* out = (float*)d_out;

    unsigned short* rn = (unsigned short*)d_ws;          // 4096x128 bf16 = 1 MB
    unsigned short* Wt = rn + (size_t)kN2 * kDout;       // 128x2048 bf16 = 512 KB

    hipMemsetAsync(d_out, 0, sizeof(float), stream);

    w_transpose_bf16<<<dim3(128), dim3(256), 0, stream>>>(W, Wt);
    simclr_encode<<<dim3(32), dim3(256), 0, stream>>>(x_i, x_j, Wt, bias, rn);
    simclr_loss_k<<<dim3(32), dim3(256), 0, stream>>>(rn, out);
}
